// ReEig_35029753266846
// MI455X (gfx1250) — matrix operand order repaired and checked by reading, not compiled, not run on hardware
//
#include <hip/hip_runtime.h>
#include <hip/hip_bf16.h>
#include <math.h>

// ReEig for 8192 symmetric 64x64 f32 matrices on gfx1250 (MI455X).
//
//   f(A) = V max(L, eta) V^T = 0.5*(A + eta*I + |A - eta*I|)
//   |M|  = sign(M) @ M,  sign(M) via Newton-Schulz with the fused form:
//          W = 1.5*I - 0.5*Y^2 ;  Y <- Y @ W
//
// All GEMMs are 64x64x64 on v_wmma_f32_16x16x32_f16 (f16 in, f32 acc).
// One matrix per workgroup; 16 wave32 waves, one 16x16 tile per wave.
// Iterates live in LDS as f16. All operands are SYMMETRIC, which buys:
//   - B fragments (column gathers) loaded as contiguous rows -> ds_load_b128
//   - C tiles stored TRANSPOSED -> one ds_store_b128 per lane (vs 8x b16)
// HBM traffic: 256 MB total -> ~11 us floor at 23.3 TB/s (memory-bound);
// ~140 GFLOP of f16 WMMA is negligible against the matrix ceiling.

typedef __attribute__((ext_vector_type(4)))  _Float16 v4h;
typedef __attribute__((ext_vector_type(8)))  _Float16 v8h;
typedef __attribute__((ext_vector_type(16))) _Float16 v16h;
typedef __attribute__((ext_vector_type(8)))  float    v8f;

#define RE_ETA   1e-4f
#define LDH      72      // f16 LDS row stride (halfs): 144 B = 9*16 B -> b128-aligned rows
#define LDF      68      // f32 LDS row stride (floats): 272 B = 17*16 B -> b128-aligned rows
#define NS_ITERS 15

// ---- ISA 7.12.2 fragment loaders (wave32), f16 LDS, all ds_load_b128 ----

// A-matrix 16x32 f16: lane L holds row M = L%16; element e maps to
// K = kb + (e/8)*16 + (L/16)*8 + (e%8)  -> two contiguous 8-half runs.
__device__ __forceinline__ v16h frag_A(const _Float16* __restrict__ S, int tm, int kstep, int lane) {
    const int m  = lane & 15;
    const int hf = lane >> 4;
    const _Float16* p = S + (tm * 16 + m) * LDH + kstep * 32 + (hf << 3);
    const v8h lo = *(const v8h*)(p);        // K = kb + hf*8  .. +7
    const v8h hi = *(const v8h*)(p + 16);   // K = kb + 16 + hf*8 .. +7
    return __builtin_shufflevector(lo, hi, 0,1,2,3,4,5,6,7,8,9,10,11,12,13,14,15);
}

// B-matrix 32x16 f16: lane L holds column N = L%16; element e maps to
// K = kb + (e/8)*16 + (L/16)*8 + (e%8). The operand is SYMMETRIC, so column n == row n
// and the gather becomes two contiguous 8-half runs.
__device__ __forceinline__ v16h frag_Bsym(const _Float16* __restrict__ S, int tn, int kstep, int lane) {
    const int n  = lane & 15;
    const int hf = lane >> 4;
    const _Float16* p = S + (tn * 16 + n) * LDH + kstep * 32 + (hf << 3);
    const v8h lo = *(const v8h*)(p);
    const v8h hi = *(const v8h*)(p + 16);
    return __builtin_shufflevector(lo, hi, 0,1,2,3,4,5,6,7,8,9,10,11,12,13,14,15);
}

__device__ __forceinline__ v8f wmma_step(v16h a, v16h b, v8f c) {
    // (neg_a, A, neg_b, B, c_mod, C, reuse_a, reuse_b)
    return __builtin_amdgcn_wmma_f32_16x16x32_f16(false, a, false, b, (short)0, c, false, false);
}

// 16x16 tile of SA @ SB over K=64 (two K=32 WMMA steps); SB symmetric.
__device__ __forceinline__ v8f gemm_tile(const _Float16* __restrict__ SA,
                                         const _Float16* __restrict__ SB,
                                         int tm, int tn, int lane) {
    v8f acc = {};
#pragma unroll
    for (int ks = 0; ks < 2; ++ks) {
        v16h a = frag_A(SA, tm, ks, lane);
        v16h b = frag_Bsym(SB, tn, ks, lane);
        acc = wmma_step(a, b, acc);
    }
    return acc;
}

// -------------------------------------------------------------------------

__global__ __launch_bounds__(512)
void reeig_ns_kernel(const float* __restrict__ X, float* __restrict__ Out) {
    __shared__ __align__(16) float    sM [64 * LDF];   // M = A - eta*I (f32, kept for output)
    __shared__ __align__(16) _Float16 sMh[64 * LDH];   // f16 mirror of M (final GEMM operand)
    __shared__ __align__(16) _Float16 sYh[64 * LDH];   // Newton-Schulz iterate (f16)
    __shared__ __align__(16) _Float16 sWh[64 * LDH];   // W = 1.5I - 0.5*Y^2 (f16)
    __shared__ float s_fro;

    const int tid  = threadIdx.x;        // 0..511 (16 wave32 waves)
    const int lane = tid & 31;
    const int wave = tid >> 5;           // 0..15
    const int tm   = wave >> 2;          // output tile row 0..3
    const int tn   = wave & 3;           // output tile col 0..3
    const int hf   = lane >> 4;          // C/D layout: M-half
    const int cn   = lane & 15;          // C/D layout: N

    // Transposed-store coordinates for this lane's C fragment:
    //   owns C[row0 .. row0+7][col]; stored (by symmetry) at [col][row0..row0+7].
    const int col  = tn * 16 + cn;
    const int row0 = tm * 16 + (hf << 3);
    const int tOffH = col * LDH + row0;  // f16 staging: 16 B aligned (LDH=72, row0%8==0)
    const int tOffF = col * LDF + row0;  // f32 staging: 16 B aligned (LDF=68, row0%4==0)

    const float*  Xb  = X   + (size_t)blockIdx.x * 4096;
    float*        Ob  = Out + (size_t)blockIdx.x * 4096;
    const float4* Xb4 = (const float4*)Xb;
    float4*       Ob4 = (float4*)Ob;

    if (tid == 0) s_fro = 0.0f;
    __syncthreads();

    // Load M = A - eta*I (b128 global loads); accumulate squared Frobenius norm.
    float part = 0.0f;
#pragma unroll
    for (int i = 0; i < 2; ++i) {
        const int idx = tid + i * 512;           // 0..1023 float4s
        const int r  = idx >> 4;                 // row
        const int c4 = (idx & 15) << 2;          // first of 4 cols
        float4 v = Xb4[idx];
        if (r == c4 + 0) v.x -= RE_ETA;
        if (r == c4 + 1) v.y -= RE_ETA;
        if (r == c4 + 2) v.z -= RE_ETA;
        if (r == c4 + 3) v.w -= RE_ETA;
        *(float4*)&sM[r * LDF + c4] = v;
        v4h h = { (_Float16)v.x, (_Float16)v.y, (_Float16)v.z, (_Float16)v.w };
        *(v4h*)&sMh[r * LDH + c4] = h;           // ds_store_b64
        part += v.x * v.x + v.y * v.y + v.z * v.z + v.w * v.w;
    }
    atomicAdd(&s_fro, part);                     // ds_add_f32
    __syncthreads();

    // Y0 = M / (||M||_F * margin): spectral radius strictly < 1.
    const float invs = 1.0f / (sqrtf(s_fro) * 1.02f + 1e-30f);
#pragma unroll
    for (int i = 0; i < 2; ++i) {
        const int idx = tid + i * 512;
        const int r  = idx >> 4;
        const int c4 = (idx & 15) << 2;
        const float4 v = *(const float4*)&sM[r * LDF + c4];
        v4h h = { (_Float16)(v.x * invs), (_Float16)(v.y * invs),
                  (_Float16)(v.z * invs), (_Float16)(v.w * invs) };
        *(v4h*)&sYh[r * LDH + c4] = h;
    }
    __syncthreads();

    // Newton-Schulz: W = 1.5I - 0.5*Y@Y ; Y <- Y @ W   (2 WMMA GEMMs / iter)
    for (int it = 0; it < NS_ITERS; ++it) {
        v8f accZ = gemm_tile(sYh, sYh, tm, tn, lane);
        {
            v8h wv;
#pragma unroll
            for (int r = 0; r < 8; ++r) {
                const float w = ((row0 + r) == col ? 1.5f : 0.0f) - 0.5f * accZ[r];
                wv[r] = (_Float16)w;
            }
            *(v8h*)&sWh[tOffH] = wv;             // transposed tile: one ds_store_b128
        }
        __syncthreads();                                   // W visible

        v8f accT = gemm_tile(sYh, sWh, tm, tn, lane);      // Y@W in regs
        __syncthreads();                                   // all reads of Y done
        {
            v8h yv;
#pragma unroll
            for (int r = 0; r < 8; ++r) yv[r] = (_Float16)accT[r];
            *(v8h*)&sYh[tOffH] = yv;             // transposed tile: one ds_store_b128
        }
        __syncthreads();                                   // new Y visible
    }

    // |M| = sign(M) @ M (scale-invariant, use unscaled f16 mirror of M).
    v8f accS = gemm_tile(sYh, sMh, tm, tn, lane);

    // Out = 0.5*M + 0.5*|M| + eta*I. Read M transposed (symmetric) as b128,
    // stage the (transposed, == untransposed) result in sM for b128 stores.
    {
        const float4 m0 = *(const float4*)&sM[tOffF];
        const float4 m1 = *(const float4*)&sM[tOffF + 4];
        float4 o0, o1;
        o0.x = 0.5f * m0.x + 0.5f * accS[0];
        o0.y = 0.5f * m0.y + 0.5f * accS[1];
        o0.z = 0.5f * m0.z + 0.5f * accS[2];
        o0.w = 0.5f * m0.w + 0.5f * accS[3];
        o1.x = 0.5f * m1.x + 0.5f * accS[4];
        o1.y = 0.5f * m1.y + 0.5f * accS[5];
        o1.z = 0.5f * m1.z + 0.5f * accS[6];
        o1.w = 0.5f * m1.w + 0.5f * accS[7];
        const int d = col - row0;                // diagonal hits at r == d
        if (d == 0) o0.x += RE_ETA;
        if (d == 1) o0.y += RE_ETA;
        if (d == 2) o0.z += RE_ETA;
        if (d == 3) o0.w += RE_ETA;
        if (d == 4) o1.x += RE_ETA;
        if (d == 5) o1.y += RE_ETA;
        if (d == 6) o1.z += RE_ETA;
        if (d == 7) o1.w += RE_ETA;
        *(float4*)&sM[tOffF]     = o0;
        *(float4*)&sM[tOffF + 4] = o1;
    }
    __syncthreads();

#pragma unroll
    for (int i = 0; i < 2; ++i) {
        const int idx = tid + i * 512;           // 0..1023 float4s
        const int r  = idx >> 4;
        const int c4 = (idx & 15) << 2;
        Ob4[idx] = *(const float4*)&sM[r * LDF + c4];   // global_store_b128
    }
}

extern "C" void kernel_launch(void* const* d_in, const int* in_sizes, int n_in,
                              void* d_out, int out_size, void* d_ws, size_t ws_size,
                              hipStream_t stream) {
    (void)n_in; (void)out_size; (void)d_ws; (void)ws_size;
    const float* x   = (const float*)d_in[0];
    float*       out = (float*)d_out;
    const int batch  = in_sizes[0] / 4096;       // 8192 matrices of 64x64
    reeig_ns_kernel<<<dim3(batch), dim3(512), 0, stream>>>(x, out);
}
